// Representation_flow_25383256720156
// MI455X (gfx1250) — compile-verified, matrix operand order checked by reading, not run on hardware
//
#include <hip/hip_runtime.h>

// ---------------------------------------------------------------------------
// Representation-flow network for MI455X (gfx1250).
// Convs run as implicit GEMM on the WMMA pipe (bf16 x bf16 -> f32, 16x16x32).
// Large GEMMs stage 128x128 macro-tiles in LDS via the Tensor Data Mover
// (TENSOR_LOAD_TO_LDS + s_wait_tensorcnt); TV-L1 runs fully LDS-resident.
// ---------------------------------------------------------------------------

typedef __attribute__((ext_vector_type(16))) __bf16 v16bf;
typedef __attribute__((ext_vector_type(8)))  float  v8f;
typedef unsigned int __attribute__((ext_vector_type(4))) v4u;
typedef int          __attribute__((ext_vector_type(8))) v8i;
typedef int          __attribute__((ext_vector_type(4))) v4i;

__device__ __forceinline__ unsigned short f2bf(float f) {
  // round-to-nearest-even f32 -> bf16
  unsigned int u = __float_as_uint(f);
  unsigned int r = u + 0x7FFFu + ((u >> 16) & 1u);
  return (unsigned short)(r >> 16);
}

// --------------------------- TDM 2D tile load ------------------------------
// Issue TENSOR_LOAD_TO_LDS for a [tileD1 rows x tileD0 elems] bf16 tile from a
// row-major tensor with row stride strideElems, remaining extent d0 x d1 for
// OOB zero-fill. Descriptor packing per CDNA5 ISA:
//   group0: [1:0]=count=1, [63:32]=lds_addr, [120:64]=global_addr, [127:126]=2
//   group1: [17:16]=data_size(1->2B), [79:48]=tensor_dim0, [111:80]=tensor_dim1,
//           [127:112]=tile_dim0, [143:128]=tile_dim1, [207:160]=dim0_stride
// This toolchain's builtin is the 6-arg form:
//   (uint32x4 g0, int32x8 g1, int32x4 g2, int32x4 g3, int32x8 g4, i32 cpol)
__device__ __forceinline__ void tdm_load_2d(const unsigned short* gbase,
                                            unsigned ldsByteOff, long elemOff,
                                            int d0, int d1, int strideElems,
                                            int tileD0, int tileD1) {
  unsigned long long ga = (unsigned long long)gbase + 2ull * (unsigned long long)elemOff;
  v4u g0;
  g0.x = 1u;                                        // count=1 (valid user D#)
  g0.y = ldsByteOff;                                // lds_addr
  g0.z = (unsigned)(ga & 0xFFFFFFFFull);            // global_addr[31:0]
  g0.w = (unsigned)((ga >> 32) & 0x01FFFFFFull)     // global_addr[56:32]
       | 0x80000000u;                               // type=2 ("image")
  v8i g1;
  g1[0] = (int)(1u << 16);                          // data_size = 2 bytes
  g1[1] = (int)(((unsigned)d0 & 0xFFFFu) << 16);    // tensor_dim0[15:0]
  g1[2] = (int)((((unsigned)d0 >> 16) & 0xFFFFu)    // tensor_dim0[31:16]
       | (((unsigned)d1 & 0xFFFFu) << 16));         // tensor_dim1[15:0]
  g1[3] = (int)((((unsigned)d1 >> 16) & 0xFFFFu)    // tensor_dim1[31:16]
       | (((unsigned)tileD0 & 0xFFFFu) << 16));     // tile_dim0
  g1[4] = (int)((unsigned)tileD1 & 0xFFFFu);        // tile_dim1 (tile_dim2=0)
  g1[5] = (int)strideElems;                         // tensor_dim0_stride[31:0]
  g1[6] = 0;                                        // stride hi / dim1 stride
  g1[7] = 0;
  v4i z4 = {0, 0, 0, 0};                            // groups 2/3 unused (2D)
  v8i z8 = {0, 0, 0, 0, 0, 0, 0, 0};                // trailing group (unused)
  __builtin_amdgcn_tensor_load_to_lds(g0, g1, z4, z4, z8, 0);
}

// ------------------------------ weight cast --------------------------------
__global__ void k_cast_bf16(const float* __restrict__ src,
                            unsigned short* __restrict__ dst, long n) {
  long i = (long)blockIdx.x * blockDim.x + threadIdx.x;
  if (i < n) dst[i] = f2bf(src[i]);
}

// ------------------------- im2col: 8x8 stride-8 ----------------------------
__global__ void k_im2col_s8(const float* __restrict__ in,
                            unsigned short* __restrict__ A,
                            int B, int Cin, int H, int W, int Ho, int Wo) {
  const int K = Cin * 64;
  long idx = (long)blockIdx.x * blockDim.x + threadIdx.x;
  long total = (long)B * Ho * Wo * K;
  if (idx >= total) return;
  int  k = (int)(idx % K);
  long m = idx / K;
  int c = k >> 6, r = k & 63, dy = r >> 3, dx = r & 7;
  int hw = Ho * Wo;
  int b = (int)(m / hw), p = (int)(m % hw);
  int oy = p / Wo, ox = p % Wo;
  float v = in[(((long)b * Cin + c) * H + oy * 8 + dy) * W + ox * 8 + dx];
  A[idx] = f2bf(v);
}

// ------------------------------ im2col: 3x3 --------------------------------
__global__ void k_im2col_3x3(const float* __restrict__ in,
                             unsigned short* __restrict__ A,
                             int B, int Cin, int H, int W, int Ho, int Wo,
                             int planar, int pad) {
  const int K = Cin * 9;
  long idx = (long)blockIdx.x * blockDim.x + threadIdx.x;
  long total = (long)B * Ho * Wo * K;
  if (idx >= total) return;
  int  k = (int)(idx % K);
  long m = idx / K;
  int c = k / 9, r = k % 9, dy = r / 3, dx = r % 3;
  int hw = Ho * Wo;
  int b = (int)(m / hw), p = (int)(m % hw);
  int oy = p / Wo, ox = p % Wo;
  int y = oy + dy - pad, x = ox + dx - pad;
  float v = 0.f;
  if (y >= 0 && y < H && x >= 0 && x < W) {
    v = planar ? in[(((long)b * Cin + c) * H + y) * W + x]
               : in[(((long)b * H + y) * W + x) * Cin + c];
  }
  A[idx] = f2bf(v);
}

// ----------------- WMMA GEMM, direct-from-global (small N) -----------------
__global__ void k_gemm_wmma(const unsigned short* __restrict__ A,
                            const unsigned short* __restrict__ Bc,
                            const float* __restrict__ bias,
                            float* __restrict__ Cf,
                            unsigned short* __restrict__ Ch,
                            int M, int N, int K, int relu) {
  long wave = ((long)blockIdx.x * blockDim.x + threadIdx.x) >> 5;
  int lane = threadIdx.x & 31;
  int nTiles = N >> 4;
  long tiles = (long)(M >> 4) * nTiles;
  if (wave >= tiles) return;                 // wave-uniform: EXEC stays all-1s
  int mTile = (int)(wave / nTiles);
  int nTile = (int)(wave % nTiles);

  int row = mTile * 16 + (lane & 15);
  int col = nTile * 16 + (lane & 15);
  int khalf = (lane >> 4) * 16;
  const unsigned short* ap = A  + (long)row * K + khalf;
  const unsigned short* bp = Bc + (long)col * K + khalf;

  v8f acc = {};
  for (int k = 0; k < K; k += 32) {
    v16bf av = *(const v16bf*)(ap + k);
    v16bf bv = *(const v16bf*)(bp + k);
    __builtin_prefetch(ap + k + 128, 0, 1);
    __builtin_prefetch(bp + k + 128, 0, 1);
    acc = __builtin_amdgcn_wmma_f32_16x16x32_bf16(
        false, av, false, bv, (short)0, acc, false, false);
  }

  float bv = bias ? bias[col] : 0.f;
  int rbase = mTile * 16 + ((lane >> 4) << 3);
  for (int v = 0; v < 8; ++v) {
    float x = acc[v] + bv;
    if (relu) x = fmaxf(x, 0.f);
    long o = (long)(rbase + v) * N + col;
    if (Cf) Cf[o] = x;
    if (Ch) Ch[o] = f2bf(x);
  }
}

// -------------- WMMA GEMM, TDM-staged 128x128 macro-tiles ------------------
// Block = 8 waves. Wave w owns rows [w*16, w*16+16) of the 128-row block and
// all (up to 8) 16-col tiles -> 8 WMMAs per 32-K step from LDS fragments.
// Wave 0 issues two TENSOR_LOAD_TO_LDS per K step (A 128x32, B nTB*16 x 32);
// TDM zero-fills rows beyond the tensor extent, so ragged M needs no masks
// in the compute loop (stores are row-guarded).
__global__ void k_gemm_wmma_tdm(const unsigned short* __restrict__ A,
                                const unsigned short* __restrict__ Bc,
                                const float* __restrict__ bias,
                                float* __restrict__ Cf,
                                unsigned short* __restrict__ Ch,
                                int M, int N, int K, int relu) {
  __shared__ unsigned short shA[128 * 32];
  __shared__ unsigned short shB[128 * 32];

  int wave = threadIdx.x >> 5;
  int lane = threadIdx.x & 31;
  int gridN = (N + 127) >> 7;
  int bm = blockIdx.x / gridN;
  int bn = blockIdx.x % gridN;
  int rowBase = bm * 128;
  int nBase = bn * 128;
  int nTB = (N - nBase) >> 4; if (nTB > 8) nTB = 8;

  unsigned ldsA = (unsigned)(unsigned long long)(void*)shA;
  unsigned ldsB = (unsigned)(unsigned long long)(void*)shB;

  v8f acc[8];
#pragma unroll
  for (int i = 0; i < 8; ++i) acc[i] = (v8f){};

  int khalf = (lane >> 4) * 16;              // lanes 0-15: K 0..15; 16-31: 16..31
  int rA = wave * 16 + (lane & 15);

  for (int kt = 0; kt < K; kt += 32) {
    __syncthreads();                         // LDS reads of prev step finished
    if (threadIdx.x < 32) {                  // one wave drives the TDM
      tdm_load_2d(A,  ldsA, (long)rowBase * K + kt,
                  K - kt, M - rowBase, K, 32, 128);
      tdm_load_2d(Bc, ldsB, (long)nBase * K + kt,
                  K - kt, N - nBase, K, 32, nTB * 16);
      __builtin_amdgcn_s_wait_tensorcnt((short)0);
    }
    __syncthreads();                         // publish LDS tile to all waves

    v16bf av = *(const v16bf*)&shA[rA * 32 + khalf];
#pragma unroll
    for (int nt = 0; nt < 8; ++nt) {
      if (nt < nTB) {                        // block-uniform: EXEC stays all-1s
        v16bf bv = *(const v16bf*)&shB[(nt * 16 + (lane & 15)) * 32 + khalf];
        acc[nt] = __builtin_amdgcn_wmma_f32_16x16x32_bf16(
            false, av, false, bv, (short)0, acc[nt], false, false);
      }
    }
  }

  int rbase = rowBase + wave * 16 + ((lane >> 4) << 3);
  for (int nt = 0; nt < nTB; ++nt) {
    int col = nBase + nt * 16 + (lane & 15);
    float bvs = bias ? bias[col] : 0.f;
#pragma unroll
    for (int v = 0; v < 8; ++v) {
      int r = rbase + v;
      if (r < M) {
        float xv = acc[nt][v] + bvs;
        if (relu) xv = fmaxf(xv, 0.f);
        long o = (long)r * N + col;
        if (Cf) Cf[o] = xv;
        if (Ch) Ch[o] = f2bf(xv);
      }
    }
  }
}

// ------------------------------- TV-L1 -------------------------------------
__global__ void k_tvl1(const float* __restrict__ I1,
                       const float* __restrict__ I2,
                       float* __restrict__ out, int C, int H, int W) {
  constexpr float THETA = 0.3f, LAMBDA = 0.15f, TAU = 0.25f, EPS = 1e-12f;
  const float l_t = LAMBDA * THETA, taut = TAU / THETA;
  const int S = H * W;
  int bc = blockIdx.x;
  int b = bc / C, c = bc % C;

  extern __shared__ float lds[];
  float* gx    = lds;
  float* gy    = gx   + S;
  float* grad  = gy   + S;
  float* rho_c = grad + S;
  float* u1    = rho_c+ S;
  float* u2    = u1   + S;
  float* p11   = u2   + S;
  float* p12   = p11  + S;
  float* p21   = p12  + S;
  float* p22   = p21  + S;

  const float* a  = I1 + (long)b * S * C + c;
  const float* bi = I2 + (long)b * S * C + c;

  for (int i = threadIdx.x; i < S; i += blockDim.x) {
    int y = i / W, x = i - y * W;
    float bxp = (x + 1 < W) ? bi[(long)(i + 1) * C] : 0.f;
    float bxm = (x > 0)     ? bi[(long)(i - 1) * C] : 0.f;
    float byp = (y + 1 < H) ? bi[(long)(i + W) * C] : 0.f;
    float bym = (y > 0)     ? bi[(long)(i - W) * C] : 0.f;
    float gxv = 0.5f * (bxp - bxm), gyv = 0.5f * (byp - bym);
    gx[i] = gxv; gy[i] = gyv;
    grad[i] = gxv * gxv + gyv * gyv + EPS;
    rho_c[i] = bi[(long)i * C] - a[(long)i * C];
    u1[i] = 0.f; u2[i] = 0.f;
    p11[i] = 0.f; p12[i] = 0.f; p21[i] = 0.f; p22[i] = 0.f;
  }
  __syncthreads();

  for (int it = 0; it < 20; ++it) {
    for (int i = threadIdx.x; i < S; i += blockDim.x) {
      int y = i / W, x = i - y * W;
      float g = grad[i];
      float rho = rho_c[i] + gx[i] * u1[i] + gy[i] * u2[i];
      float v1, v2;
      if (rho < -l_t * g)      { v1 = u1[i] + l_t * gx[i]; v2 = u2[i] + l_t * gy[i]; }
      else if (rho > l_t * g)  { v1 = u1[i] - l_t * gx[i]; v2 = u2[i] - l_t * gy[i]; }
      else { float d = -rho / g; v1 = u1[i] + d * gx[i];   v2 = u2[i] + d * gy[i]; }
      float d1 = (p11[i] - (x > 0 ? p11[i - 1] : 0.f))
               + (p12[i] - (y > 0 ? p12[i - W] : 0.f));
      float d2 = (p21[i] - (x > 0 ? p21[i - 1] : 0.f))
               + (p22[i] - (y > 0 ? p22[i - W] : 0.f));
      u1[i] = v1 + THETA * d1;
      u2[i] = v2 + THETA * d2;
    }
    __syncthreads();
    for (int i = threadIdx.x; i < S; i += blockDim.x) {
      int y = i / W, x = i - y * W;
      float u1x = (x + 1 < W) ? u1[i + 1] - u1[i] : 0.f;
      float u1y = (y + 1 < H) ? u1[i + W] - u1[i] : 0.f;
      float u2x = (x + 1 < W) ? u2[i + 1] - u2[i] : 0.f;
      float u2y = (y + 1 < H) ? u2[i + W] - u2[i] : 0.f;
      float n1 = 1.f + taut * sqrtf(u1x * u1x + u1y * u1y + EPS);
      float n2 = 1.f + taut * sqrtf(u2x * u2x + u2y * u2y + EPS);
      p11[i] = (p11[i] + taut * u1x) / n1;
      p12[i] = (p12[i] + taut * u1y) / n1;
      p21[i] = (p21[i] + taut * u2x) / n2;
      p22[i] = (p22[i] + taut * u2y) / n2;
    }
    __syncthreads();
  }

  float* o1 = out + ((long)b * 2 * C + c) * S;
  float* o2 = out + ((long)b * 2 * C + C + c) * S;
  for (int i = threadIdx.x; i < S; i += blockDim.x) {
    o1[i] = u1[i];
    o2[i] = u2[i];
  }
}

// ------------------------------ pool + FC ----------------------------------
__global__ void k_pool(const float* __restrict__ e, float* __restrict__ pooled,
                       int B, int N, int S) {
  int idx = blockIdx.x * blockDim.x + threadIdx.x;
  if (idx >= B * N) return;
  int b = idx / N, n = idx % N;
  const float* p = e + (long)b * S * N + n;
  float s = 0.f;
  for (int i = 0; i < S; ++i) s += p[(long)i * N];
  pooled[idx] = s / (float)S;
}

__global__ void k_fc(const float* __restrict__ pooled,
                     const float* __restrict__ Wfc,
                     const float* __restrict__ bfc,
                     float* __restrict__ outp, int B, int Nin, int Nout) {
  int idx = blockIdx.x * blockDim.x + threadIdx.x;
  if (idx >= B * Nout) return;
  int b = idx / Nout, n = idx % Nout;
  float s = bfc[n];
  const float* pr = pooled + (long)b * Nin;
  const float* wr = Wfc + (long)n * Nin;
  for (int k = 0; k < Nin; ++k) s += pr[k] * wr[k];
  outp[idx] = s;
}

// ---------------------------------------------------------------------------
static inline int cdiv(long a, long b) { return (int)((a + b - 1) / b); }

static void gemm_small(hipStream_t s, const unsigned short* A, const unsigned short* Bc,
                       const float* bias, float* Cf, unsigned short* Ch,
                       int M, int N, int K, int relu) {
  long tiles = (long)(M / 16) * (N / 16);
  int blocks = cdiv(tiles, 8);
  k_gemm_wmma<<<blocks, 256, 0, s>>>(A, Bc, bias, Cf, Ch, M, N, K, relu);
}

static void gemm_tdm(hipStream_t s, const unsigned short* A, const unsigned short* Bc,
                     const float* bias, float* Cf, unsigned short* Ch,
                     int M, int N, int K, int relu) {
  int blocks = cdiv(M, 128) * cdiv(N, 128);
  k_gemm_wmma_tdm<<<blocks, 256, 0, s>>>(A, Bc, bias, Cf, Ch, M, N, K, relu);
}

extern "C" void kernel_launch(void* const* d_in, const int* in_sizes, int n_in,
                              void* d_out, int out_size, void* d_ws, size_t ws_size,
                              hipStream_t stream) {
  (void)in_sizes; (void)n_in; (void)out_size; (void)ws_size;

  const float* x[4] = {(const float*)d_in[0], (const float*)d_in[1],
                       (const float*)d_in[2], (const float*)d_in[3]};
  const float* w_start = (const float*)d_in[4];
  const float* b_start = (const float*)d_in[5];
  const float* w_red   = (const float*)d_in[6];
  const float* b_red   = (const float*)d_in[7];
  const float* w_flow  = (const float*)d_in[8];
  const float* b_flow  = (const float*)d_in[9];
  const float* w_exp   = (const float*)d_in[10];
  const float* b_exp   = (const float*)d_in[11];
  const float* w_end   = (const float*)d_in[12];
  const float* b_end   = (const float*)d_in[13];
  const float* w_fc    = (const float*)d_in[14];
  const float* b_fc    = (const float*)d_in[15];

  const int Bn = 8;
  const int H1 = 56, S1 = H1 * H1, M1 = Bn * S1;   // 25088
  const int K1 = 192;
  const int H2 = 54, S2 = H2 * H2, M2 = Bn * S2;   // 23328
  const int Kf = 576;
  const int H3 = 52, S3 = H3 * H3, M3 = Bn * S3;   // 21632
  const int Ke = 1152;

  char* ws = (char*)d_ws;
  size_t off = 0;
  auto alloc = [&](size_t bytes) -> void* {
    void* p = ws + off;
    off = (off + bytes + 255) & ~(size_t)255;
    return p;
  };
  unsigned short* wbS   = (unsigned short*)alloc((size_t)128 * K1 * 2);
  unsigned short* wbR   = (unsigned short*)alloc((size_t)32 * 128 * 2);
  unsigned short* wbF   = (unsigned short*)alloc((size_t)64 * Kf * 2);
  unsigned short* wbE   = (unsigned short*)alloc((size_t)128 * Ke * 2);
  unsigned short* wbEnd = (unsigned short*)alloc((size_t)512 * Ke * 2);
  unsigned short* A1    = (unsigned short*)alloc((size_t)M1 * K1 * 2);
  unsigned short* featH = (unsigned short*)alloc((size_t)M1 * 128 * 2);
  float* r[4];
  for (int i = 0; i < 4; ++i) r[i] = (float*)alloc((size_t)M1 * 32 * 4);
  float* f11 = (float*)alloc((size_t)Bn * 64 * S1 * 4);
  float* f12 = (float*)alloc((size_t)Bn * 64 * S1 * 4);
  unsigned short* A2 = (unsigned short*)alloc((size_t)M2 * Kf * 2);
  float* cf1 = (float*)alloc((size_t)M2 * 64 * 4);
  float* cf2 = (float*)alloc((size_t)M2 * 64 * 4);
  float* f2  = (float*)alloc((size_t)Bn * 128 * S2 * 4);
  unsigned short* A3 = (unsigned short*)alloc((size_t)M3 * Ke * 2);
  float* o3  = (float*)alloc((size_t)M3 * 128 * 4);
  float* e   = (float*)alloc((size_t)M3 * 512 * 4);
  float* pooled = (float*)alloc((size_t)Bn * 512 * 4);

  k_cast_bf16<<<cdiv((long)128 * K1, 256), 256, 0, stream>>>(w_start, wbS, (long)128 * K1);
  k_cast_bf16<<<cdiv((long)32 * 128, 256), 256, 0, stream>>>(w_red, wbR, (long)32 * 128);
  k_cast_bf16<<<cdiv((long)64 * Kf, 256), 256, 0, stream>>>(w_flow, wbF, (long)64 * Kf);
  k_cast_bf16<<<cdiv((long)128 * Ke, 256), 256, 0, stream>>>(w_exp, wbE, (long)128 * Ke);
  k_cast_bf16<<<cdiv((long)512 * Ke, 256), 256, 0, stream>>>(w_end, wbEnd, (long)512 * Ke);

  // feat(x_i): 8x8/s8 conv + relu (bf16 out), then 1x1 conv 128->32 (f32 out)
  for (int i = 0; i < 4; ++i) {
    k_im2col_s8<<<cdiv((long)M1 * K1, 256), 256, 0, stream>>>(
        x[i], A1, Bn, 3, 448, 448, H1, H1);
    gemm_tdm(stream, A1, wbS, b_start, nullptr, featH, M1, 128, K1, 1);
    gemm_small(stream, featH, wbR, b_red, r[i], nullptr, M1, 32, 128, 0);
  }

  // TV-L1 stage 1 (C=32, 56x56): 125440B of LDS per workgroup
  size_t lds1 = (size_t)10 * S1 * 4;
  k_tvl1<<<Bn * 32, 256, lds1, stream>>>(r[0], r[1], f11, 32, H1, H1);
  k_tvl1<<<Bn * 32, 256, lds1, stream>>>(r[2], r[3], f12, 32, H1, H1);

  // flow conv 3x3 VALID 64->64 on both flow fields
  k_im2col_3x3<<<cdiv((long)M2 * Kf, 256), 256, 0, stream>>>(
      f11, A2, Bn, 64, H1, H1, H2, H2, 1, 0);
  gemm_tdm(stream, A2, wbF, b_flow, cf1, nullptr, M2, 64, Kf, 0);
  k_im2col_3x3<<<cdiv((long)M2 * Kf, 256), 256, 0, stream>>>(
      f12, A2, Bn, 64, H1, H1, H2, H2, 1, 0);
  gemm_tdm(stream, A2, wbF, b_flow, cf2, nullptr, M2, 64, Kf, 0);

  // TV-L1 stage 2 (C=64, 54x54)
  size_t lds2 = (size_t)10 * S2 * 4;
  k_tvl1<<<Bn * 64, 256, lds2, stream>>>(cf1, cf2, f2, 64, H2, H2);

  // expand conv 3x3 VALID 128->128
  k_im2col_3x3<<<cdiv((long)M3 * Ke, 256), 256, 0, stream>>>(
      f2, A3, Bn, 128, H2, H2, H3, H3, 1, 0);
  gemm_tdm(stream, A3, wbE, b_exp, o3, nullptr, M3, 128, Ke, 0);

  // end conv 3x3 SAME 128->512 + relu
  k_im2col_3x3<<<cdiv((long)M3 * Ke, 256), 256, 0, stream>>>(
      o3, A3, Bn, 128, H3, H3, H3, H3, 0, 1);
  gemm_tdm(stream, A3, wbEnd, b_end, e, nullptr, M3, 512, Ke, 1);

  // global mean pool + FC head
  k_pool<<<cdiv(Bn * 512, 256), 256, 0, stream>>>(e, pooled, Bn, 512, S3);
  k_fc<<<cdiv(Bn * 400, 256), 256, 0, stream>>>(pooled, w_fc, b_fc,
                                                (float*)d_out, Bn, 512, 400);
}
